// BertAttention_44968307589137
// MI455X (gfx1250) — compile-verified, matrix operand order checked by reading, not run on hardware
//
#include <hip/hip_runtime.h>

// Problem constants (BERT attention): B=4, S=2048, HID=1024, NH=16, HD=64
#define CB   4
#define CS   2048
#define CHID 1024
#define CNH  16
#define CHD  64
#define LROW (CS + 4)   // padded f32 LDS row stride (floats)
#define PROW (CS + 8)   // padded f16 LDS row stride (halves)

typedef __attribute__((ext_vector_type(16))) _Float16 v16h;
typedef __attribute__((ext_vector_type(8)))  _Float16 v8h;
typedef __attribute__((ext_vector_type(8)))  float    v8f;
typedef __attribute__((ext_vector_type(4)))  float    v4f;
typedef __attribute__((ext_vector_type(4)))  int      v4i;

// CDNA5 WMMA: D(16x16 f32) = A(16x32 f16) * B(32x16 f16) + C
__device__ __forceinline__ v8f wmma16(v16h a, v16h b, v8f c) {
  return __builtin_amdgcn_wmma_f32_16x16x32_f16(
      /*neg_a=*/false, a, /*neg_b=*/false, b,
      /*c_mod=*/(short)0, c, /*reuse_a=*/false, /*reuse_b=*/false);
}

// A/B fragment loader from f16 row-major storage (global or LDS).
// Per ISA 16-bit A-matrix 16x32 layout: lane L holds row (L&15);
// halves [0..7]  = K = kb .. kb+7,  halves [8..15] = K = kb+16 .. kb+23,
// where kb = (L>>4)*8. B(32x16) is symmetric: lane holds column (L&15).
__device__ __forceinline__ v16h frag_f16(const _Float16* base, int kb) {
  v8h lo = *(const v8h*)(base + kb);
  v8h hi = *(const v8h*)(base + kb + 16);
  return __builtin_shufflevector(lo, hi,
      0, 1, 2, 3, 4, 5, 6, 7, 8, 9, 10, 11, 12, 13, 14, 15);
}

// Same fragment layout, sourced from fp32 memory with on-the-fly cvt.
__device__ __forceinline__ v16h frag_f32(const float* base, int kb) {
  v4f a0 = *(const v4f*)(base + kb);
  v4f a1 = *(const v4f*)(base + kb + 4);
  v4f b0 = *(const v4f*)(base + kb + 16);
  v4f b1 = *(const v4f*)(base + kb + 20);
  v16h r;
#pragma unroll
  for (int i = 0; i < 4; ++i) {
    r[i]      = (_Float16)a0[i];
    r[4 + i]  = (_Float16)a1[i];
    r[8 + i]  = (_Float16)b0[i];
    r[12 + i] = (_Float16)b1[i];
  }
  return r;
}

// ---------------------------------------------------------------------------
// Kernel 1: projection  out = act @ W^T + bias, emitted as f16.
// Each wave computes a 16-token x 64-col (= one head) tile.
// transposeV==0: out[b][h][s][d]   (Q, K)
// transposeV==1: out[b][h][d][s]   (V, transposed for contiguous ctx B-frags)
// ---------------------------------------------------------------------------
__global__ void qkv_proj_kernel(const float* __restrict__ act,
                                const float* __restrict__ W,
                                const float* __restrict__ bias,
                                _Float16* __restrict__ out,
                                int transposeV) {
  const int lane  = threadIdx.x & 31;
  const int wave  = threadIdx.x >> 5;
  const int mtile = blockIdx.x * 4 + wave;   // 0..511 (8192 tokens / 16)
  const int h     = blockIdx.y;              // head == 64-col tile, 0..15
  const int mbase = mtile * 16;
  const int row   = lane & 15;
  const int kb    = (lane >> 4) * 8;

  const float* arow = act + (size_t)(mbase + row) * CHID;

  v8f acc[4];
#pragma unroll
  for (int t = 0; t < 4; ++t) acc[t] = (v8f){0.f,0.f,0.f,0.f,0.f,0.f,0.f,0.f};

  for (int kc = 0; kc < CHID; kc += 32) {
    v16h a = frag_f32(arow + kc, kb);
#pragma unroll
    for (int t = 0; t < 4; ++t) {
      const float* wrow = W + (size_t)(h * 64 + t * 16 + row) * CHID + kc;
      v16h bm = frag_f32(wrow, kb);
      acc[t] = wmma16(a, bm, acc[t]);
    }
  }

  // C/D layout: element i of v8f holds M = i + 8*(lane>>4), N = lane&15.
  const int Mo = (lane >> 4) * 8;
  const int N  = lane & 15;
#pragma unroll
  for (int t = 0; t < 4; ++t) {
    const int col = h * 64 + t * 16 + N;   // global output column
    const int d   = t * 16 + N;            // dim within head
    const float bv = bias[col];
#pragma unroll
    for (int i = 0; i < 8; ++i) {
      const int g = mbase + Mo + i;        // global token
      const int b = g >> 11;               // / S
      const int s = g & (CS - 1);
      const float val = acc[t][i] + bv;
      if (!transposeV)
        out[(((size_t)(b * CNH + h)) * CS + s) * CHD + d] = (_Float16)val;
      else
        out[(((size_t)(b * CNH + h)) * CHD + d) * CS + s] = (_Float16)val;
    }
  }
}

// ---------------------------------------------------------------------------
// Kernel 2: fused attention for one (b,h) and one 16-row q-tile per block.
// 4 waves / 128 threads.
// LDS: 16 x LROW fp32 scores (128 KB) + 16 x PROW f16 probs (64 KB).
//   phase 1 : raw scores = QK^T * 0.125 via WMMA -> f32 LDS
//   phase 1b: coalesced mask (int4) + masked-score writeback; NT float4
//             stream of scores to d_out
//   phase 2 : row softmax (wave32 shuffle reductions); probs -> f16 LDS
//   phase 3 : ctx = probs @ V via WMMA (A-frags = ds_load_b128 from f16 LDS)
// ---------------------------------------------------------------------------
__global__ void attn_fused_kernel(const _Float16* __restrict__ Q16,
                                  const _Float16* __restrict__ K16,
                                  const _Float16* __restrict__ VT16,
                                  const int* __restrict__ amask,
                                  float* __restrict__ ctx_out,
                                  float* __restrict__ scores_out) {
  extern __shared__ char smem_raw[];
  float*    lds  = (float*)smem_raw;                                  // scores
  _Float16* plds = (_Float16*)(smem_raw + (size_t)16 * LROW * 4);     // probs

  const int lane  = threadIdx.x & 31;
  const int wave  = threadIdx.x >> 5;      // 0..3
  const int bh    = blockIdx.x;            // 0..63
  const int b     = bh >> 4;
  const int h     = bh & 15;
  const int qbase = blockIdx.y * 16;

  const _Float16* Qhead = Q16  + (size_t)bh * CS * CHD;
  const _Float16* Khead = K16  + (size_t)bh * CS * CHD;
  const _Float16* Vhead = VT16 + (size_t)bh * CHD * CS;

  const int row = lane & 15;
  const int kb  = (lane >> 4) * 8;
  const int Mo  = (lane >> 4) * 8;
  const int N   = lane & 15;

  // Q-tile A fragments, K-dim 0..31 and 32..63 (resident for whole phase 1)
  const _Float16* qrow = Qhead + (size_t)(qbase + row) * CHD;
  v16h qa0 = frag_f16(qrow, kb);
  v16h qa1 = frag_f16(qrow + 32, kb);

  // ---- phase 1: score tiles -> LDS, striped over waves -------------------
  for (int kt = wave; kt < CS / 16; kt += 4) {
    const int kbase0 = kt * 16;
    const _Float16* krow = Khead + (size_t)(kbase0 + row) * CHD;
    if (kt + 4 < CS / 16)
      __builtin_prefetch(Khead + (size_t)((kt + 4) * 16 + row) * CHD, 0, 1);

    v16h kb0 = frag_f16(krow, kb);
    v16h kb1 = frag_f16(krow + 32, kb);
    v8f c = (v8f){0.f,0.f,0.f,0.f,0.f,0.f,0.f,0.f};
    c = wmma16(qa0, kb0, c);
    c = wmma16(qa1, kb1, c);

#pragma unroll
    for (int i = 0; i < 8; ++i)
      lds[(Mo + i) * LROW + kbase0 + N] = c[i] * 0.125f;   // 1/sqrt(64)
  }
  __syncthreads();

  // ---- phase 1b: coalesced mask apply + NT stream of scores --------------
  {
    const int*  mrow = amask      + ((size_t)b  * CS + qbase) * CS;
    float*      srow = scores_out + ((size_t)bh * CS + qbase) * CS;
    for (int idx = threadIdx.x; idx < 16 * (CS / 4); idx += 128) {
      const int r = idx >> 9;              // / (CS/4)
      const int c = (idx & (CS / 4 - 1)) * 4;
      v4f sv = *(const v4f*)(lds + r * LROW + c);
      v4i mv = *(const v4i*)(mrow + (size_t)r * CS + c);
      v4f o;
#pragma unroll
      for (int i = 0; i < 4; ++i) o[i] = (mv[i] == 0) ? -1.0e9f : sv[i];
      *(v4f*)(lds + r * LROW + c) = o;
      __builtin_nontemporal_store(o, (v4f*)(srow + (size_t)r * CS + c));
    }
  }
  __syncthreads();

  // ---- phase 2: softmax, 4 rows per wave; probs emitted as f16 -----------
  for (int r = wave * 4; r < wave * 4 + 4; ++r) {
    float*    rowp  = lds  + r * LROW;
    _Float16* prow  = plds + r * PROW;
    float m = -3.0e38f;
    for (int k = lane; k < CS; k += 32) m = fmaxf(m, rowp[k]);
#pragma unroll
    for (int off = 16; off; off >>= 1) m = fmaxf(m, __shfl_xor(m, off, 32));
    float sum = 0.f;
    for (int k = lane; k < CS; k += 32) {
      float p = __expf(rowp[k] - m);
      sum += p;
      rowp[k] = p;
    }
#pragma unroll
    for (int off = 16; off; off >>= 1) sum += __shfl_xor(sum, off, 32);
    const float inv = 1.0f / sum;
    for (int k = lane; k < CS; k += 32)
      prow[k] = (_Float16)(rowp[k] * inv);
  }
  __syncthreads();

  // ---- phase 3: ctx = probs @ V, wave w owns d-cols [16w, 16w+16) --------
  const int dbase = wave * 16;
  v8f acc = (v8f){0.f,0.f,0.f,0.f,0.f,0.f,0.f,0.f};
  for (int kc = 0; kc < CS; kc += 32) {
    // A fragment: f16 probs straight from LDS (two ds_load_b128)
    v16h a = frag_f16(plds + row * PROW + kc, kb);
    // B fragment: column d = dbase + (lane&15) of V == contiguous row of V^T
    v16h bm = frag_f16(Vhead + (size_t)(dbase + row) * CS + kc, kb);
    acc = wmma16(a, bm, acc);
  }

  // ctx[b][s][h*64 + d], s = qbase+M, d = dbase+N
#pragma unroll
  for (int i = 0; i < 8; ++i) {
    const int M = Mo + i;
    ctx_out[((size_t)b * CS + qbase + M) * CHID + h * 64 + dbase + N] = acc[i];
  }
}

// ---------------------------------------------------------------------------
extern "C" void kernel_launch(void* const* d_in, const int* in_sizes, int n_in,
                              void* d_out, int out_size, void* d_ws, size_t ws_size,
                              hipStream_t stream) {
  (void)in_sizes; (void)n_in; (void)out_size; (void)ws_size;

  const float* query = (const float*)d_in[0];
  const float* key   = (const float*)d_in[1];
  const float* value = (const float*)d_in[2];
  const int*   amask = (const int*)d_in[3];
  const float* Wq    = (const float*)d_in[4];
  const float* bq    = (const float*)d_in[5];
  const float* Wk    = (const float*)d_in[6];
  const float* bk    = (const float*)d_in[7];
  const float* Wv    = (const float*)d_in[8];
  const float* bv    = (const float*)d_in[9];

  float* ctx    = (float*)d_out;                       // (B,S,HID)
  float* scores = ctx + (size_t)CB * CS * CHID;        // (B,NH,S,S)

  const size_t per = (size_t)CB * CNH * CS * CHD;      // elems per f16 tensor
  _Float16* Q16  = (_Float16*)d_ws;
  _Float16* K16  = Q16 + per;
  _Float16* VT16 = Q16 + 2 * per;

  // projections: 512 m-tiles -> 128 blocks of 4 waves; grid.y = 16 heads
  dim3 pblk(128, 1, 1);
  dim3 pgrd(128, 16, 1);
  qkv_proj_kernel<<<pgrd, pblk, 0, stream>>>(query, Wq, bq, Q16, 0);
  qkv_proj_kernel<<<pgrd, pblk, 0, stream>>>(key,   Wk, bk, K16, 0);
  qkv_proj_kernel<<<pgrd, pblk, 0, stream>>>(value, Wv, bv, VT16, 1);

  // fused attention: 197 KB LDS per workgroup (WGP has 320 KB)
  const size_t smem = (size_t)16 * LROW * sizeof(float)
                    + (size_t)16 * PROW * sizeof(_Float16);
  (void)hipFuncSetAttribute((const void*)attn_fused_kernel,
                            hipFuncAttributeMaxDynamicSharedMemorySize,
                            (int)smem);
  dim3 ablk(128, 1, 1);
  dim3 agrd(CB * CNH, CS / 16, 1);
  attn_fused_kernel<<<agrd, ablk, smem, stream>>>(Q16, K16, VT16, amask,
                                                  ctx, scores);
}
